// Attention1D_82506321756715
// MI455X (gfx1250) — matrix operand order repaired and checked by reading, not compiled, not run on hardware
//
#include <hip/hip_runtime.h>
#include <hip/hip_bf16.h>

#define DEVI __device__ __forceinline__

typedef __attribute__((ext_vector_type(4)))  float  f32x4;
typedef __attribute__((ext_vector_type(8)))  float  v8f;
typedef __attribute__((ext_vector_type(16))) __bf16 v16bf;
typedef __attribute__((ext_vector_type(8)))  __bf16 bf16x8;

union BFrag { v16bf v; f32x4 q[2]; };
union BPack { bf16x8 h; f32x4 q; };

constexpr int B_ = 16, C_ = 256, L_ = 2048;

// workspace layout (bytes)
constexpr size_t OFF_WQ = 0;
constexpr size_t OFF_WK = 131072;
constexpr size_t OFF_WV = 262144;
constexpr size_t OFF_WO = 393216;
constexpr size_t OFF_HT = 524288;                       // hT  [b][l][c] bf16
constexpr size_t SZ_T   = (size_t)B_ * L_ * C_ * 2;     // 16 MB per tensor
constexpr size_t OFF_QT = OFF_HT + SZ_T;                // qT  [b][l][c]
constexpr size_t OFF_KT = OFF_QT + SZ_T;                // kT  [b][l][c]
constexpr size_t OFF_VV = OFF_KT + SZ_T;                // v   [b][c][l]
constexpr size_t OFF_OT = OFF_VV + SZ_T;                // oT  [b][l][c]

// flash-kernel dynamic LDS partition (elements)
constexpr int KS_SZ = 32 * 264;        // one K tile
constexpr int VS_SZ = 256 * 40;        // one V tile
constexpr int QS_E = 0;                // Qs [128][264]
constexpr int KS_E = 128 * 264;        // Ks double buffer
constexpr int VS_E = KS_E + 2 * KS_SZ; // Vs double buffer
constexpr int PS_E = VS_E + 2 * VS_SZ; // Ps [8][16][40]
constexpr unsigned FLASH_LDS = (PS_E + 8 * 16 * 40) * 2;   // 152576 B

DEVI v8f wmma_bf16(v16bf a, v16bf b, v8f c) {
    return __builtin_amdgcn_wmma_f32_16x16x32_bf16(false, a, false, b,
                                                   (short)0, c, false, false);
}

// Async global -> LDS, 16 bytes per lane, tracked by ASYNCcnt.
// VDST = per-lane LDS byte address; generic LDS pointers truncate to the
// hardware LDS offset (ISA 10.2).
DEVI void async_b128(const __bf16* lds_dst, const __bf16* gsrc) {
    unsigned loff = (unsigned)(unsigned long long)(uintptr_t)lds_dst;
    unsigned long long ga = (unsigned long long)(uintptr_t)gsrc;
    asm volatile("global_load_async_to_lds_b128 %0, %1, off"
                 :: "v"(loff), "v"(ga) : "memory");
}

DEVI void wait_async0() {
#if __has_builtin(__builtin_amdgcn_s_wait_asynccnt)
    __builtin_amdgcn_s_wait_asynccnt(0);
#else
    asm volatile("s_wait_asynccnt 0x0" ::: "memory");
#endif
}

// A-matrix fragment (16x32 bf16, M rows x K cols), row-major source, stride ld.
// ISA layout: lanes 0-15 hold M=lane, K{0..7,16..23}; lanes 16-31 K{8..15,24..31}.
DEVI v16bf lds_frag_a(const __bf16* base, int ld, int koff, int lane) {
    int m = lane & 15, h = (lane >> 4) & 1;
    BFrag f;
    f.q[0] = *(const f32x4*)(base + m * ld + koff + h * 8);
    f.q[1] = *(const f32x4*)(base + m * ld + koff + 16 + h * 8);
    return f.v;
}

// B-matrix fragment (32x16 bf16, K rows x N cols) from a transposed tile
// Bt[n][k] (stride ld). Lanes 0-15 hold column n=lane, K{0..7,16..23}; lanes 16-31 K{8..15,24..31}.
DEVI v16bf lds_frag_b(const __bf16* base, int ld, int koff, int lane) {
    int n = lane & 15, h = (lane >> 4) & 1;
    BFrag f;
    f.q[0] = *(const f32x4*)(base + n * ld + koff + h * 8);
    f.q[1] = *(const f32x4*)(base + n * ld + koff + 16 + h * 8);
    return f.v;
}

// ---------------- weight conversion f32 -> bf16 ----------------
__global__ __launch_bounds__(256) void cvt_w(const float* __restrict__ qw,
                                             const float* __restrict__ kw,
                                             const float* __restrict__ vw,
                                             const float* __restrict__ ow,
                                             __bf16* __restrict__ dst) {
    int i = blockIdx.x * 256 + threadIdx.x;   // 0 .. 4*65536-1
    int m = i >> 16, e = i & 65535;
    float v;
    if      (m == 0) v = qw[e];
    else if (m == 1) v = kw[e];
    else if (m == 2) v = vw[e];
    else             v = ow[e];
    dst[i] = (__bf16)v;
}

// ---------------- GroupNorm: x[b,c,l] -> hT[b,l,c] bf16 ----------------
__global__ __launch_bounds__(256) void gnorm(const float* __restrict__ x,
                                             const float* __restrict__ w,
                                             const float* __restrict__ bb,
                                             __bf16* __restrict__ hT) {
    int g = blockIdx.x, b = blockIdx.y, t = threadIdx.x;
    int c0 = g * 8;
    const float* xb = x + ((size_t)b * C_ + c0) * L_;

    float s = 0.f, s2 = 0.f;
    for (int ci = 0; ci < 8; ++ci)
        for (int it = 0; it < 8; ++it) {
            float v = xb[ci * L_ + it * 256 + t];
            s += v; s2 += v * v;
        }
    for (int m = 16; m; m >>= 1) {
        s  += __shfl_xor(s,  m, 32);
        s2 += __shfl_xor(s2, m, 32);
    }
    __shared__ float rs[8], rs2[8], stats[2];
    int wv = t >> 5, ln = t & 31;
    if (!ln) { rs[wv] = s; rs2[wv] = s2; }
    __syncthreads();
    if (t == 0) {
        float a = 0.f, a2 = 0.f;
        for (int i = 0; i < 8; ++i) { a += rs[i]; a2 += rs2[i]; }
        float mean = a * (1.f / 16384.f);
        float var  = a2 * (1.f / 16384.f) - mean * mean;
        stats[0] = mean;
        stats[1] = rsqrtf(var + 1e-5f);
    }
    __syncthreads();
    float mean = stats[0], inv = stats[1];
    float gw[8], gb[8];
    #pragma unroll
    for (int ci = 0; ci < 8; ++ci) {
        float ww = w[c0 + ci];
        gw[ci] = ww * inv;
        gb[ci] = bb[c0 + ci] - mean * inv * ww;
    }
    for (int it = 0; it < 8; ++it) {
        int l = it * 256 + t;
        BPack p;
        #pragma unroll
        for (int ci = 0; ci < 8; ++ci) {
            float v = xb[ci * L_ + l];
            p.h[ci] = (__bf16)(v * gw[ci] + gb[ci]);
        }
        *(f32x4*)(hT + ((size_t)b * L_ + l) * C_ + c0) = p.q;  // 16B store
    }
}

// ---------------- generic channel-projection GEMM ----------------
// D[o,l] = sum_c W[o,c] * Ht[b,l,c] (+bias).  MODE 0: store bf16 transposed
// [b,l,o]; MODE 1: store bf16 [b,o,l]; MODE 2: f32 [b,o,l] + bias + residual.
// Staging is async global->LDS, double buffered over ASYNCcnt.
template <int MODE>
__global__ __launch_bounds__(256, 1) void proj(const __bf16* __restrict__ W,
                                               const float*  __restrict__ bias,
                                               const __bf16* __restrict__ Ht,
                                               __bf16* __restrict__ outb,
                                               float*  __restrict__ outf,
                                               const float* __restrict__ xres) {
    __shared__ __bf16 As[2][64 * 40];    // W tile   64(M) x 32(K)
    __shared__ __bf16 Bt[2][128 * 40];   // hT tile 128(N) x 32(K)

    int t = threadIdx.x, lane = t & 31, wave = t >> 5;
    int b = blockIdx.z, l0 = blockIdx.x * 128, o0 = blockIdx.y * 64;
    int wm = wave & 1, wn = wave >> 1;   // 2 x 4 wave grid, 32x32 per wave

    v8f acc[2][2] = {};
    const __bf16* hb = Ht + ((size_t)b * L_ + l0) * C_;

    int rowA = t >> 2, segA = t & 3;     // A staging map
    int nB = t >> 1, hfB = t & 1;        // B staging map

    // prologue: async-stage k-step 0 into buffer 0
    async_b128(As[0] + rowA * 40 + segA * 8, W + (o0 + rowA) * C_ + segA * 8);
    {
        const __bf16* src = hb + nB * C_ + hfB * 16;
        __bf16* bd = Bt[0] + nB * 40 + hfB * 16;
        async_b128(bd,     src);
        async_b128(bd + 8, src + 8);
    }
    wait_async0();
    __syncthreads();

    #pragma unroll 1
    for (int step = 0; step < 8; ++step) {
        int buf = step & 1;
        if (step < 7) {           // async-stage next k-step into other buffer
            int kn = (step + 1) * 32;
            async_b128(As[buf ^ 1] + rowA * 40 + segA * 8,
                       W + (o0 + rowA) * C_ + kn + segA * 8);
            const __bf16* src = hb + nB * C_ + kn + hfB * 16;
            __bf16* bd = Bt[buf ^ 1] + nB * 40 + hfB * 16;
            async_b128(bd,     src);
            async_b128(bd + 8, src + 8);
        }
        v16bf af[2], bf[2];
        #pragma unroll
        for (int i = 0; i < 2; ++i)
            af[i] = lds_frag_a(As[buf] + (wm * 32 + i * 16) * 40, 40, 0, lane);
        #pragma unroll
        for (int j = 0; j < 2; ++j)
            bf[j] = lds_frag_b(Bt[buf] + (wn * 32 + j * 16) * 40, 40, 0, lane);
        #pragma unroll
        for (int i = 0; i < 2; ++i)
            #pragma unroll
            for (int j = 0; j < 2; ++j)
                acc[i][j] = wmma_bf16(af[i], bf[j], acc[i][j]);
        wait_async0();
        __syncthreads();
    }

    int hf = lane >> 4, nn = lane & 15;
    #pragma unroll
    for (int i = 0; i < 2; ++i) {
        int obase = o0 + wm * 32 + i * 16 + hf * 8;
        #pragma unroll
        for (int j = 0; j < 2; ++j) {
            int l = l0 + wn * 32 + j * 16 + nn;
            if (MODE == 0) {              // bf16, [b][l][o]
                BPack p;
                #pragma unroll
                for (int r = 0; r < 8; ++r)
                    p.h[r] = (__bf16)(acc[i][j][r] + bias[obase + r]);
                *(f32x4*)(outb + ((size_t)b * L_ + l) * C_ + obase) = p.q;
            } else if (MODE == 1) {       // bf16, [b][o][l]
                #pragma unroll
                for (int r = 0; r < 8; ++r)
                    outb[((size_t)b * C_ + obase + r) * L_ + l] =
                        (__bf16)(acc[i][j][r] + bias[obase + r]);
            } else {                      // f32 out + bias + residual
                #pragma unroll
                for (int r = 0; r < 8; ++r) {
                    size_t idx = ((size_t)b * C_ + obase + r) * L_ + l;
                    outf[idx] = acc[i][j][r] + bias[obase + r] + xres[idx];
                }
            }
        }
    }
}

// ---------------- fused flash attention ----------------
// qT,kT: [b][l][c] bf16; v: [b][c][l] bf16; out oT: [b][l][c] bf16.
// Dynamic LDS (152.6 KB): Q block + double-buffered K/V tiles (async-staged)
// + per-wave P bounce.
__global__ __launch_bounds__(256, 1) void flash(const __bf16* __restrict__ Qt,
                                                const __bf16* __restrict__ Kt,
                                                const __bf16* __restrict__ Vv,
                                                __bf16* __restrict__ Ot) {
    extern __shared__ __bf16 smem[];
    __bf16* Qs = smem + QS_E;          // [128][264]
    __bf16* Ks = smem + KS_E;          // 2 x [32][264]
    __bf16* Vs = smem + VS_E;          // 2 x [256][40]
    __bf16* Ps = smem + PS_E;          // [8][16][40]

    int t = threadIdx.x, lane = t & 31, wave = t >> 5;
    int b = blockIdx.y;
    int lblk = blockIdx.x * 128;
    int l0 = lblk + wave * 16;             // 16 query rows per wave
    int mrow = lane & 15, hf = lane >> 4;
    int rowK = t >> 3, segK = t & 7;       // K staging map

    const __bf16* kb = Kt + (size_t)b * L_ * C_;
    const __bf16* vb = Vv + (size_t)b * C_ * L_;

    // async-stage whole Q block (128 x 256): 2 threads per row, 256B each
    {
        int row = t >> 1, part = t & 1;
        const __bf16* src = Qt + ((size_t)b * L_ + lblk + row) * C_ + part * 128;
        __bf16* dst = Qs + row * 264 + part * 128;
        #pragma unroll
        for (int q = 0; q < 16; ++q) async_b128(dst + q * 8, src + q * 8);
    }
    // async-stage first K/V tile into buffer 0
    {
        const __bf16* src = kb + (size_t)rowK * C_ + segK * 32;
        __bf16* dst = Ks + rowK * 264 + segK * 32;
        #pragma unroll
        for (int q = 0; q < 4; ++q) async_b128(dst + q * 8, src + q * 8);
        const __bf16* vsrc = vb + (size_t)t * L_;
        __bf16* vdst = Vs + t * 40;
        #pragma unroll
        for (int q = 0; q < 4; ++q) async_b128(vdst + q * 8, vsrc + q * 8);
    }

    v8f accO[16] = {};
    float mrun[8], lrun[8];
    #pragma unroll
    for (int r = 0; r < 8; ++r) { mrun[r] = -1e30f; lrun[r] = 0.f; }

    const __bf16* Qw = Qs + wave * 16 * 264;
    __bf16* Pw = Ps + wave * 16 * 40;

    wait_async0();
    __syncthreads();   // Q block + first K/V tile visible to all waves

    int buf = 0;
    for (int mt = 0; mt < 64; ++mt) {
        if (mt < 63) {   // async-stage next K/V tile into the other buffer
            int mn = (mt + 1) * 32;
            const __bf16* src = kb + (size_t)(mn + rowK) * C_ + segK * 32;
            __bf16* dst = Ks + (buf ^ 1) * KS_SZ + rowK * 264 + segK * 32;
            #pragma unroll
            for (int q = 0; q < 4; ++q) async_b128(dst + q * 8, src + q * 8);
            const __bf16* vsrc = vb + (size_t)t * L_ + mn;
            __bf16* vdst = Vs + (buf ^ 1) * VS_SZ + t * 40;
            #pragma unroll
            for (int q = 0; q < 4; ++q) async_b128(vdst + q * 8, vsrc + q * 8);
        }
        const __bf16* Kc = Ks + buf * KS_SZ;
        const __bf16* Vc = Vs + buf * VS_SZ;

        // S = (Q K^T) * scale : 16(l) x 32(m), K-dim = 256.
        // distance-1 software pipeline of LDS fragment loads
        v8f s0 = {}, s1 = {};
        v16bf aC  = lds_frag_a(Qw, 264, 0, lane);
        v16bf b0C = lds_frag_b(Kc, 264, 0, lane);
        v16bf b1C = lds_frag_b(Kc + 16 * 264, 264, 0, lane);
        #pragma unroll
        for (int kc = 0; kc < 8; ++kc) {
            v16bf aN = aC, b0N = b0C, b1N = b1C;
            if (kc < 7) {
                aN  = lds_frag_a(Qw, 264, (kc + 1) * 32, lane);
                b0N = lds_frag_b(Kc, 264, (kc + 1) * 32, lane);
                b1N = lds_frag_b(Kc + 16 * 264, 264, (kc + 1) * 32, lane);
            }
            s0 = wmma_bf16(aC, b0C, s0);
            s1 = wmma_bf16(aC, b1C, s1);
            aC = aN; b0C = b0N; b1C = b1N;
        }

        float p0[8], p1[8], alpha[8];
        #pragma unroll
        for (int r = 0; r < 8; ++r) {
            p0[r] = s0[r] * 0.0625f;   // C^-0.5
            p1[r] = s1[r] * 0.0625f;
        }
        // online softmax per row (rows live in 16-lane halves of the wave)
        #pragma unroll
        for (int r = 0; r < 8; ++r) {
            float mx = fmaxf(p0[r], p1[r]);
            for (int mk = 8; mk; mk >>= 1) mx = fmaxf(mx, __shfl_xor(mx, mk, 32));
            float nm = fmaxf(mrun[r], mx);
            alpha[r] = __expf(mrun[r] - nm);
            mrun[r]  = nm;
            p0[r] = __expf(p0[r] - nm);
            p1[r] = __expf(p1[r] - nm);
            float rsum = p0[r] + p1[r];
            for (int mk = 8; mk; mk >>= 1) rsum += __shfl_xor(rsum, mk, 32);
            lrun[r] = lrun[r] * alpha[r] + rsum;
        }
        #pragma unroll
        for (int ch = 0; ch < 16; ++ch)
            #pragma unroll
            for (int r = 0; r < 8; ++r)
                accO[ch][r] = accO[ch][r] * alpha[r];

        // redistribute P: D-layout -> A-layout via per-wave LDS tile [l][m]
        #pragma unroll
        for (int r = 0; r < 8; ++r) {
            Pw[(r + 8 * hf) * 40 + mrow]      = (__bf16)p0[r];
            Pw[(r + 8 * hf) * 40 + 16 + mrow] = (__bf16)p1[r];
        }
        v16bf pa = lds_frag_a(Pw, 40, 0, lane);

        // O += P V^T : A = P (16 x 32), B chunks over c, pipelined loads
        v16bf bvC = lds_frag_b(Vc, 40, 0, lane);
        #pragma unroll
        for (int ch = 0; ch < 16; ++ch) {
            v16bf bvN = bvC;
            if (ch < 15) bvN = lds_frag_b(Vc + (ch + 1) * 16 * 40, 40, 0, lane);
            accO[ch] = wmma_bf16(pa, bvC, accO[ch]);
            bvC = bvN;
        }
        wait_async0();
        __syncthreads();
        buf ^= 1;
    }

    float inv[8];
    #pragma unroll
    for (int r = 0; r < 8; ++r) inv[r] = 1.f / lrun[r];

    // write oT[b][l][c] bf16, coalescing each 16x16 chunk through LDS
    #pragma unroll
    for (int ch = 0; ch < 16; ++ch) {
        #pragma unroll
        for (int r = 0; r < 8; ++r)
            Pw[(r + 8 * hf) * 40 + mrow] = (__bf16)(accO[ch][r] * inv[r]);
        BPack p;
        p.q = *(const f32x4*)(Pw + mrow * 40 + hf * 8);
        *(f32x4*)(Ot + ((size_t)b * L_ + l0 + mrow) * C_ + ch * 16 + hf * 8) = p.q;
    }
}

extern "C" void kernel_launch(void* const* d_in, const int* in_sizes, int n_in,
                              void* d_out, int out_size, void* d_ws, size_t ws_size,
                              hipStream_t stream) {
    const float* x   = (const float*)d_in[0];
    const float* nw  = (const float*)d_in[1];
    const float* nb  = (const float*)d_in[2];
    const float* qw  = (const float*)d_in[3];
    const float* qb  = (const float*)d_in[4];
    const float* kw  = (const float*)d_in[5];
    const float* kb  = (const float*)d_in[6];
    const float* vw  = (const float*)d_in[7];
    const float* vb  = (const float*)d_in[8];
    const float* ow  = (const float*)d_in[9];
    const float* ob  = (const float*)d_in[10];
    float* out = (float*)d_out;

    char* ws = (char*)d_ws;
    __bf16* WB = (__bf16*)(ws);
    __bf16* WQ = (__bf16*)(ws + OFF_WQ);
    __bf16* WK = (__bf16*)(ws + OFF_WK);
    __bf16* WV = (__bf16*)(ws + OFF_WV);
    __bf16* WO = (__bf16*)(ws + OFF_WO);
    __bf16* HT = (__bf16*)(ws + OFF_HT);
    __bf16* QT = (__bf16*)(ws + OFF_QT);
    __bf16* KT = (__bf16*)(ws + OFF_KT);
    __bf16* VV = (__bf16*)(ws + OFF_VV);
    __bf16* OT = (__bf16*)(ws + OFF_OT);

    (void)hipFuncSetAttribute((const void*)flash,
                              hipFuncAttributeMaxDynamicSharedMemorySize,
                              (int)FLASH_LDS);

    cvt_w<<<1024, 256, 0, stream>>>(qw, kw, vw, ow, WB);
    gnorm<<<dim3(32, 16), 256, 0, stream>>>(x, nw, nb, HT);

    dim3 pg(16, 4, 16);   // N-tiles(128) x M-tiles(64) x batch
    proj<0><<<pg, 256, 0, stream>>>(WQ, qb, HT, QT, nullptr, nullptr);
    proj<0><<<pg, 256, 0, stream>>>(WK, kb, HT, KT, nullptr, nullptr);
    proj<1><<<pg, 256, 0, stream>>>(WV, vb, HT, VV, nullptr, nullptr);

    flash<<<dim3(16, 16), 256, FLASH_LDS, stream>>>(QT, KT, VV, OT);

    proj<2><<<pg, 256, 0, stream>>>(WO, ob, OT, nullptr, out, x);
}
